// AsyncCKCNN_71459665871145
// MI455X (gfx1250) — matrix operand order repaired and checked by reading, not compiled, not run on hardware
//
#include <hip/hip_runtime.h>
#include <hip/hip_bf16.h>

// ---------------------------------------------------------------------------
// Problem constants (from reference): B=32, N=512, P=257, Cin=32, H=64, D=16
// ---------------------------------------------------------------------------
#define BB   32
#define NN   512
#define PP   257
#define PPAD 272      // 17 * 16 (M-tiles)
#define CIN  32
#define HH   64
#define DD   16
#define KTOT 544      // Cin * (D+1) = 17 WMMA k-steps of 32
#define W0C  30.0f
#define SLOPE 0.01f   // jax.nn.leaky_relu default

typedef _Float16 v16h __attribute__((ext_vector_type(16)));
typedef float    v8f  __attribute__((ext_vector_type(8)));

union FragH { uint4 q[2]; v16h v; };

// ---------------------------------------------------------------------------
// K0: build WmatT[h][c*17+d] f16 : fused W3r / b3r, transposed so each WMMA
// B-fragment lane reads two 16-byte runs of contiguous k.
// ---------------------------------------------------------------------------
__global__ void k0_prep(const float* __restrict__ w3, const float* __restrict__ b3,
                        _Float16* __restrict__ WmatT) {
    int i = blockIdx.x * 256 + threadIdx.x;         // 64*544
    if (i >= HH * KTOT) return;
    int h = i / KTOT, k = i % KTOT;
    int c = k / 17, d = k % 17;
    float v = (d < DD) ? w3[d * (HH * CIN) + h * CIN + c] : b3[h * CIN + c];
    WmatT[i] = (_Float16)v;
}

// ---------------------------------------------------------------------------
// K3: generate conv kernel kf[s][o*64+c] = SIREN MLP on rel2=s/256 (f16)
// ---------------------------------------------------------------------------
__global__ void k3_gen(const float* __restrict__ w1, const float* __restrict__ b1,
                       const float* __restrict__ w2, const float* __restrict__ b2,
                       const float* __restrict__ w3, const float* __restrict__ b3,
                       _Float16* __restrict__ kf) {
    int s = blockIdx.x, tid = threadIdx.x;
    float rel = (float)s * (1.0f / 256.0f);
    float h1v[DD], h2v[DD];
#pragma unroll
    for (int d = 0; d < DD; ++d) h1v[d] = __sinf(W0C * (rel * w1[d] + b1[d]));
#pragma unroll
    for (int d = 0; d < DD; ++d) {
        float a = b2[d];
#pragma unroll
        for (int e = 0; e < DD; ++e) a += h1v[e] * w2[e * DD + d];
        h2v[d] = __sinf(W0C * a);
    }
    int j0 = tid * 16;                               // 256 threads * 16 = 4096
    for (int j = j0; j < j0 + 16; ++j) {
        float a = b3[j];
#pragma unroll
        for (int d = 0; d < DD; ++d) a += h2v[d] * w3[d * (HH * HH) + j];
        kf[(size_t)s * (HH * HH) + j] = (_Float16)a;
    }
}

// ---------------------------------------------------------------------------
// K1: class-bucketed accumulation  Saug[b][p][c*17+d] = sum_{n:feat=c} w*h[d]
//     row c*17+16 accumulates sum w  (folds in the b3 term).
//     SIREN MLP on VALU (v_sin_f32), scatter via ds_add_f32 LDS atomics.
// ---------------------------------------------------------------------------
__global__ void k1_scatter(const float* __restrict__ x, const float* __restrict__ positions,
                           const float* __restrict__ w1, const float* __restrict__ b1,
                           const float* __restrict__ w2, const float* __restrict__ b2,
                           _Float16* __restrict__ Saug) {
    __shared__ float S[4 * KTOT];
    __shared__ float ts[NN], vs[NN];
    __shared__ int   cs[NN];
    __shared__ float lw1[DD], lb1[DD], lw2[DD * DD], lb2[DD];

    const int b = blockIdx.y, p0 = blockIdx.x * 4, tid = threadIdx.x;
    for (int i = tid; i < 4 * KTOT; i += 256) S[i] = 0.f;
    for (int n = tid; n < NN; n += 256) {
        float f = x[((size_t)b * NN + n) * 3 + 0];
        float v = x[((size_t)b * NN + n) * 3 + 1];
        float t = x[((size_t)b * NN + n) * 3 + 2];
        bool valid = (f != 0.f) || (v != 0.f) || (t != 0.f);
        ts[n] = t; cs[n] = (int)f; vs[n] = valid ? v : 0.f;
    }
    if (tid < DD) { lw1[tid] = w1[tid]; lb1[tid] = b1[tid]; lb2[tid] = b2[tid]; }
    lw2[tid] = w2[tid];                              // blockDim == 256 == D*D
    __syncthreads();

    const float tmax = positions[PP - 1];
    for (int pp = 0; pp < 4; ++pp) {
        int p = p0 + pp;
        if (p >= PP) continue;
        float posp = positions[p];
        for (int n = tid; n < NN; n += 256) {
            float w = vs[n];
            if (w == 0.f) continue;
            float rel = (ts[n] - posp) / tmax;
            if (rel > 0.f) continue;                 // causal mask
            float hv[DD];
#pragma unroll
            for (int d = 0; d < DD; ++d) hv[d] = __sinf(W0C * (rel * lw1[d] + lb1[d]));
            float* Sp = &S[pp * KTOT + cs[n] * 17];
#pragma unroll
            for (int d = 0; d < DD; ++d) {
                float a = lb2[d];
#pragma unroll
                for (int e = 0; e < DD; ++e) a += hv[e] * lw2[e * DD + d];
                atomicAdd(&Sp[d], w * __sinf(W0C * a));   // ds_add_f32
            }
            atomicAdd(&Sp[DD], w);
        }
    }
    __syncthreads();
    for (int i = tid; i < 4 * KTOT; i += 256) {
        int row = i / KTOT, k = i % KTOT;
        Saug[((size_t)b * PPAD + (p0 + row)) * KTOT + k] = (_Float16)S[i];
    }
}

// ---------------------------------------------------------------------------
// K2: h1[b][p][h] = Saug[b] (272x544) @ WmatT^T (544x64) + bias1  via WMMA.
//     4 waves / block, one 16-wide h-tile per wave, 17 k-steps of K=32.
//     Epilogue stores are unconditional into the 272-row padded buffer.
// ---------------------------------------------------------------------------
__global__ void k2_h1gemm(const _Float16* __restrict__ Saug, const _Float16* __restrict__ WmatT,
                          const float* __restrict__ bias1, float* __restrict__ h1) {
    const int b = blockIdx.y, p0 = blockIdx.x * 16;
    const int wave = threadIdx.x >> 5, lane = threadIdx.x & 31;
    const int m = lane & 15, hi = lane >> 4;

    v8f acc = {};
    const _Float16* arow = Saug + ((size_t)b * PPAD + (p0 + m)) * KTOT + (hi ? 8 : 0);
    const _Float16* brow = WmatT + (size_t)(wave * 16 + m) * KTOT + (hi ? 8 : 0);
#pragma unroll 1
    for (int kk = 0; kk < 17; ++kk) {
        FragH A, B;
        const uint4* ap = (const uint4*)(arow + kk * 32);
        A.q[0] = ap[0];                              // K = klo..klo+7
        A.q[1] = ap[2];                              // K = klo+16..klo+23
        const uint4* bp = (const uint4*)(brow + kk * 32);
        B.q[0] = bp[0];                              // K = klo..klo+7
        B.q[1] = bp[2];
        acc = __builtin_amdgcn_wmma_f32_16x16x32_f16(false, A.v, false, B.v,
                                                     (short)0, acc, false, false);
    }
    const int h = wave * 16 + m;
    const float bia = bias1[h];
    float* dst = h1 + ((size_t)b * PPAD + p0 + hi * 8) * HH + h;   // rows <= 271
#pragma unroll
    for (int r = 0; r < 8; ++r) dst[r * HH] = acc[r] + bia;
}

// ---------------------------------------------------------------------------
// GN1: per-batch mean/var over (H,P), normalize + leaky, emit transposed
//      zero-padded f16 activations h1nT[b][pp=p+16][h] (288 rows) for the conv.
// ---------------------------------------------------------------------------
__global__ void k_gn1(const float* __restrict__ h1, const float* __restrict__ g,
                      const float* __restrict__ bb, float* __restrict__ stats,
                      _Float16* __restrict__ h1nT) {
    const int b = blockIdx.x, tid = threadIdx.x;
    __shared__ float r0[256], r1[256];
    __shared__ float smu, srs;
    const float* hb = h1 + (size_t)b * PPAD * HH;
    float s0 = 0.f, s1 = 0.f;
    for (int i = tid; i < PP * HH; i += 256) { float v = hb[i]; s0 += v; s1 += v * v; }
    r0[tid] = s0; r1[tid] = s1; __syncthreads();
    for (int st = 128; st > 0; st >>= 1) {
        if (tid < st) { r0[tid] += r0[tid + st]; r1[tid] += r1[tid + st]; }
        __syncthreads();
    }
    if (tid == 0) {
        float m = r0[0] * (1.f / (PP * HH));
        float var = r1[0] * (1.f / (PP * HH)) - m * m;
        smu = m; srs = rsqrtf(var + 1e-12f);
        stats[b * 2] = smu; stats[b * 2 + 1] = srs;
    }
    __syncthreads();
    _Float16* out = h1nT + (size_t)b * 288 * HH;
    for (int i = tid; i < 288 * HH; i += 256) {
        int pp = i >> 6, h = i & 63;
        float v = 0.f;
        if (pp >= 16 && pp < 16 + PP) {
            float xv = hb[(pp - 16) * HH + h];
            v = (xv - smu) * srs * g[h] + bb[h];
            v = v > 0.f ? v : SLOPE * v;
        }
        out[i] = (_Float16)v;
    }
}

// ---------------------------------------------------------------------------
// K4: causal conv  h2[b][p][o] = sum_{s<=p} h1n[b][:, p-s] . kf[s][o][:]
//     WMMA GEMM, two p-tiles per block sharing each kf B-fragment:
//     4 WMMAs per B load in the main loop (halves kf L2 traffic).
//     A = shifted activation rows from a 41KB LDS tile (16-row zero pad).
// ---------------------------------------------------------------------------
__global__ void k4_conv(const _Float16* __restrict__ h1nT, const _Float16* __restrict__ kf,
                        const float* __restrict__ bias2, float* __restrict__ h2) {
    __shared__ uint sh[288 * 36];                    // 288 rows x 64 f16, stride 72 f16
    const int b = blockIdx.y, p0 = blockIdx.x * 32, tid = threadIdx.x;

    const uint* src = (const uint*)(h1nT + (size_t)b * 288 * HH);
    for (int i = tid; i < 288 * 32; i += 128) {
        int r = i >> 5, c = i & 31;
        sh[r * 36 + c] = src[i];
    }
    __syncthreads();

    const int wave = tid >> 5, lane = tid & 31;
    const int m = lane & 15, hi = lane >> 4;
    const int o = wave * 16 + m;
    const int koff = hi * 8;
    const bool has2 = (p0 + 16) < PPAD;              // blockIdx.x < 8

    v8f acc1 = {}, acc2 = {};
    const int smax1 = min(PP - 1, p0 + 15);

    // main loop: both p-tiles share B fragments
    for (int s = 0; s <= smax1; ++s) {
        const int pp1 = p0 + m + 16 - s;
        const uint4* arow1 = (const uint4*)&sh[pp1 * 36];
        const _Float16* brow = kf + ((size_t)s * HH + o) * HH + koff;
        __builtin_prefetch(brow + HH * HH, 0, 1);    // next lag (L2-resident)
        FragH A0, A1, B0, B1;
        const uint4* bq = (const uint4*)brow;
        B0.q[0] = bq[0]; B0.q[1] = bq[2];            // c 0..31
        B1.q[0] = bq[4]; B1.q[1] = bq[6];            // c 32..63
        A0.q[0] = arow1[hi];     A0.q[1] = arow1[hi + 2];
        A1.q[0] = arow1[hi + 4]; A1.q[1] = arow1[hi + 6];
        acc1 = __builtin_amdgcn_wmma_f32_16x16x32_f16(false, A0.v, false, B0.v,
                                                      (short)0, acc1, false, false);
        acc1 = __builtin_amdgcn_wmma_f32_16x16x32_f16(false, A1.v, false, B1.v,
                                                      (short)0, acc1, false, false);
        if (has2) {
            const uint4* arow2 = (const uint4*)&sh[(pp1 + 16) * 36];
            FragH C0, C1;
            C0.q[0] = arow2[hi];     C0.q[1] = arow2[hi + 2];
            C1.q[0] = arow2[hi + 4]; C1.q[1] = arow2[hi + 6];
            acc2 = __builtin_amdgcn_wmma_f32_16x16x32_f16(false, C0.v, false, B0.v,
                                                          (short)0, acc2, false, false);
            acc2 = __builtin_amdgcn_wmma_f32_16x16x32_f16(false, C1.v, false, B1.v,
                                                          (short)0, acc2, false, false);
        }
    }
    // tail: lags that only touch the second p-tile
    if (has2) {
        const int smax2 = min(PP - 1, p0 + 31);
        for (int s = p0 + 16; s <= smax2; ++s) {
            const int pp2 = p0 + 16 + m + 16 - s;
            const uint4* arow2 = (const uint4*)&sh[pp2 * 36];
            const _Float16* brow = kf + ((size_t)s * HH + o) * HH + koff;
            FragH C0, C1, B0, B1;
            const uint4* bq = (const uint4*)brow;
            B0.q[0] = bq[0]; B0.q[1] = bq[2];
            B1.q[0] = bq[4]; B1.q[1] = bq[6];
            C0.q[0] = arow2[hi];     C0.q[1] = arow2[hi + 2];
            C1.q[0] = arow2[hi + 4]; C1.q[1] = arow2[hi + 6];
            acc2 = __builtin_amdgcn_wmma_f32_16x16x32_f16(false, C0.v, false, B0.v,
                                                          (short)0, acc2, false, false);
            acc2 = __builtin_amdgcn_wmma_f32_16x16x32_f16(false, C1.v, false, B1.v,
                                                          (short)0, acc2, false, false);
        }
    }

    const float bia = bias2[o];
    float* dst = h2 + ((size_t)b * PPAD + p0 + hi * 8) * HH + o;   // rows <= 271
#pragma unroll
    for (int r = 0; r < 8; ++r) dst[r * HH] = acc1[r] + bia;
    if (has2) {
        float* dst2 = dst + 16 * HH;
#pragma unroll
        for (int r = 0; r < 8; ++r) dst2[r * HH] = acc2[r] + bia;
    }
}

// ---------------------------------------------------------------------------
// K5: GN2 stats over full h2, then z only at p=P-1 (residual with re-derived
//     post-GN1 h1), leaky, 32x64 linear -> d_out[b][o].
// ---------------------------------------------------------------------------
__global__ void k5_final(const float* __restrict__ h2, const float* __restrict__ h1,
                         const float* __restrict__ stats,
                         const float* __restrict__ g1, const float* __restrict__ b1v,
                         const float* __restrict__ g2, const float* __restrict__ b2v,
                         const float* __restrict__ lw, const float* __restrict__ lb,
                         float* __restrict__ out) {
    const int b = blockIdx.x, tid = threadIdx.x;
    __shared__ float r0[256], r1[256];
    __shared__ float zb[HH];
    __shared__ float smu, srs;
    const float* hb = h2 + (size_t)b * PPAD * HH;
    float s0 = 0.f, s1 = 0.f;
    for (int i = tid; i < PP * HH; i += 256) { float v = hb[i]; s0 += v; s1 += v * v; }
    r0[tid] = s0; r1[tid] = s1; __syncthreads();
    for (int st = 128; st > 0; st >>= 1) {
        if (tid < st) { r0[tid] += r0[tid + st]; r1[tid] += r1[tid + st]; }
        __syncthreads();
    }
    if (tid == 0) {
        float m = r0[0] * (1.f / (PP * HH));
        float var = r1[0] * (1.f / (PP * HH)) - m * m;
        smu = m; srs = rsqrtf(var + 1e-12f);
    }
    __syncthreads();
    if (tid < HH) {
        int h = tid;
        float raw = hb[(PP - 1) * HH + h];
        float gn2 = (raw - smu) * srs * g2[h] + b2v[h];
        float x1 = h1[((size_t)b * PPAD + (PP - 1)) * HH + h];
        float m1 = stats[b * 2], rs1 = stats[b * 2 + 1];
        float n1 = (x1 - m1) * rs1 * g1[h] + b1v[h];
        n1 = n1 > 0.f ? n1 : SLOPE * n1;
        float z = gn2 + n1;
        zb[h] = z > 0.f ? z : SLOPE * z;
    }
    __syncthreads();
    if (tid < 32) {
        float a = lb[tid];
#pragma unroll
        for (int h = 0; h < HH; ++h) a += lw[tid * HH + h] * zb[h];
        out[b * 32 + tid] = a;
    }
}

// ---------------------------------------------------------------------------
extern "C" void kernel_launch(void* const* d_in, const int* in_sizes, int n_in,
                              void* d_out, int out_size, void* d_ws, size_t ws_size,
                              hipStream_t stream) {
    const float* x     = (const float*)d_in[0];
    const float* pos   = (const float*)d_in[1];
    const float* k1w1  = (const float*)d_in[2];
    const float* k1b1  = (const float*)d_in[3];
    const float* k1w2  = (const float*)d_in[4];
    const float* k1b2  = (const float*)d_in[5];
    const float* k1w3  = (const float*)d_in[6];
    const float* k1b3  = (const float*)d_in[7];
    const float* bias1 = (const float*)d_in[8];
    const float* k2w1  = (const float*)d_in[9];
    const float* k2b1  = (const float*)d_in[10];
    const float* k2w2  = (const float*)d_in[11];
    const float* k2b2  = (const float*)d_in[12];
    const float* k2w3  = (const float*)d_in[13];
    const float* k2b3  = (const float*)d_in[14];
    const float* bias2 = (const float*)d_in[15];
    const float* ln1g  = (const float*)d_in[16];
    const float* ln1b  = (const float*)d_in[17];
    const float* ln2g  = (const float*)d_in[18];
    const float* ln2b  = (const float*)d_in[19];
    const float* linw  = (const float*)d_in[20];
    const float* linb  = (const float*)d_in[21];
    float* out = (float*)d_out;

    char* w = (char*)d_ws;
    auto alloc = [&](size_t bytes) -> void* {
        void* p = (void*)w;
        w += (bytes + 255) & ~(size_t)255;
        return p;
    };
    _Float16* WmatT = (_Float16*)alloc((size_t)HH * KTOT * 2);
    _Float16* Saug  = (_Float16*)alloc((size_t)BB * PPAD * KTOT * 2);
    float*    h1    = (float*)   alloc((size_t)BB * PPAD * HH * 4);
    float*    stats = (float*)   alloc((size_t)BB * 2 * 4);
    _Float16* h1nT  = (_Float16*)alloc((size_t)BB * 288 * HH * 2);
    _Float16* kf    = (_Float16*)alloc((size_t)PP * HH * HH * 2);
    float*    h2    = (float*)   alloc((size_t)BB * PPAD * HH * 4);

    k0_prep   <<<(HH * KTOT + 255) / 256, 256, 0, stream>>>(k1w3, k1b3, WmatT);
    k3_gen    <<<PP, 256, 0, stream>>>(k2w1, k2b1, k2w2, k2b2, k2w3, k2b3, kf);
    k1_scatter<<<dim3(PPAD / 4, BB), 256, 0, stream>>>(x, pos, k1w1, k1b1, k1w2, k1b2, Saug);
    k2_h1gemm <<<dim3(PPAD / 16, BB), 128, 0, stream>>>(Saug, WmatT, bias1, h1);
    k_gn1     <<<BB, 256, 0, stream>>>(h1, ln1g, ln1b, stats, h1nT);
    k4_conv   <<<dim3(9, BB), 128, 0, stream>>>(h1nT, kf, bias2, h2);
    k5_final  <<<BB, 256, 0, stream>>>(h2, h1, stats, ln1g, ln1b, ln2g, ln2b, linw, linb, out);
}
